// HopfRNNCellTheta_80221399154935
// MI455X (gfx1250) — matrix operand order repaired and checked by reading, not compiled, not run on hardware
//
#include <hip/hip_runtime.h>
#include <hip/hip_bf16.h>

typedef __attribute__((ext_vector_type(16))) __bf16 v16bf;
typedef __attribute__((ext_vector_type(8)))  __bf16 v8bf;
typedef __attribute__((ext_vector_type(8)))  float  v8f;

constexpr int kU = 1024;   // units
constexpr int kT = 512;    // time steps
constexpr int kB = 64;     // batch
constexpr int kBlocks = 32;                     // persistent grid: 32 blocks x 8 waves = 256 tiles
constexpr float kOmega = 0.00613592315154256f;  // 2*pi/1024
constexpr float kMu    = 1.0f;

// ---------------------------------------------------------------------------
// Pack f32 [U,U] row-major weights into bf16 WMMA B-operand tile layout.
// Tile grid: kt in [0,32) over K (contraction), nt in [0,64) over N.
// Within a 32x16 tile, lane l owns 16 contiguous bf16 at offset lane*16:
//   element j -> W[K = kt*32 + 8*(l>>4) + j + 8*(j>>3), N = nt*16 + (l&15)]
// Matrices packed back-to-back: [Ar, Ai, Br, Bi], 2048 tiles * 512 elems each.
// ---------------------------------------------------------------------------
__global__ void pack_weights_kernel(const float* __restrict__ Ar,
                                    const float* __restrict__ Ai,
                                    const float* __restrict__ Br,
                                    const float* __restrict__ Bi,
                                    __bf16* __restrict__ dst) {
  const int gid  = blockIdx.x * blockDim.x + threadIdx.x;  // 4*2048*32 threads
  const int lane = gid & 31;
  const int tile = (gid >> 5) & 2047;   // kt*64 + nt
  const int mat  = gid >> 16;           // 0..3
  const float* W = (mat == 0) ? Ar : (mat == 1) ? Ai : (mat == 2) ? Br : Bi;
  const int nt = tile & 63;
  const int kt = tile >> 6;
  const int kbase = (kt << 5) + ((lane >> 4) << 3);
  const int n = (nt << 4) + (lane & 15);
  __bf16* out = dst + (((size_t)mat * 2048 + tile) << 9) + (lane << 4);
#pragma unroll
  for (int j = 0; j < 16; ++j) {
    out[j] = (__bf16)W[(size_t)(kbase + j + ((j >> 3) << 3)) * kU + n];
  }
}

// z state lives as bf16 row-major [kB][kU] (rounded once per step; identical to
// what the WMMA would consume anyway). Also resets the inter-step barrier.
__global__ void init_state_kernel(const float* __restrict__ z0r,
                                  const float* __restrict__ z0i,
                                  __bf16* __restrict__ zr,
                                  __bf16* __restrict__ zi,
                                  unsigned* __restrict__ bar) {
  const int i = blockIdx.x * blockDim.x + threadIdx.x;
  if (i < kB * kU) { zr[i] = (__bf16)z0r[i]; zi[i] = (__bf16)z0i[i]; }
  if (i == 0) { *bar = 0u; }  // graph-replay safe
}

// ---------------------------------------------------------------------------
// Persistent recurrent kernel: loops t = 0..T-1 entirely on-device.
// Per step:   z_new = 0.5*(z@A + x_t@B);   y_t = z_new * ((1+mu-|z|^2) + i*w)
// Wave-per-tile: 4 (batch) x 64 (units) = 256 tiles = 32 blocks * 8 waves.
// Complex GEMM via 4 shared f32 accumulators over both matmuls:
//   rr = Re(a)@Re(W), ii = Im(a)@Im(W), ri = Re(a)@Im(W), ir = Im(a)@Re(W)
//   re = 0.5*(rr - ii), im = 0.5*(ri + ir)
// z state is bf16 row-major; ping-pong pointers are derived by arithmetic on
// the zst kernarg (no pointer array) so address-space inference keeps all
// state traffic on the GLOBAL path (LOADcnt/STOREcnt only, no FLAT/DScnt).
// Steps separated by a device-scope release/acquire atomic barrier
// (monotonic counter: target = 32*(t+1)), replacing 512 kernel launches.
// ---------------------------------------------------------------------------
__global__ __launch_bounds__(256)
void hopf_persistent_kernel(const float* __restrict__ xr,
                            const float* __restrict__ xi,
                            const __bf16* __restrict__ pW,
                            __bf16* __restrict__ zst,   // [2 buffers][re/im][kB*kU]
                            float* __restrict__ out,
                            unsigned* __restrict__ bar) {
  const int lane  = threadIdx.x & 31;
  const int wid   = (blockIdx.x << 3) + (threadIdx.x >> 5);
  const int tileM = wid & 3;          // batch tile (0..3)
  const int tileN = wid >> 2;         // unit tile  (0..63)
  const int row   = (tileM << 4) + (lane & 15);   // batch row 0..63
  const int koff  = (lane >> 4) << 3;             // 0 or 8 (A-operand lane split)

  const __bf16* pAr = pW;
  const __bf16* pAi = pW + ((size_t)1 << 20);
  const __bf16* pBr = pW + ((size_t)2 << 20);
  const __bf16* pBi = pW + ((size_t)3 << 20);

  constexpr size_t kPlane = (size_t)kB * kU;   // one re/im plane
  constexpr size_t kHalf  = 2 * kPlane;        // one ping-pong buffer (re+im)

  for (int t = 0; t < kT; ++t) {
    const size_t p = (size_t)(t & 1);
    // Pure arithmetic on zst: keeps GLOBAL address-space provenance.
    const __bf16* zr_row = zst + p * kHalf + (size_t)row * kU;
    const __bf16* zi_row = zr_row + kPlane;
    __bf16* zr_out = zst + (p ^ 1) * kHalf;
    __bf16* zi_out = zr_out + kPlane;
    const float* xr_row = xr + ((size_t)row * kT + t) * kU;
    const float* xi_row = xi + ((size_t)row * kT + t) * kU;

    v8f rr = {}, ii = {}, ri = {}, ir = {};

    // Prefetch next timestep's x rows into cache while this step computes.
    if (t + 1 < kT) {
      __builtin_prefetch(xr_row + kU + lane * 32, 0, 0);
      __builtin_prefetch(xi_row + kU + lane * 32, 0, 0);
    }

    // ---- recurrent term: z @ A  (bf16 state, no conversions) ----
    for (int kb = 0; kb < 32; ++kb) {
      const int k0 = (kb << 5) + koff;
      const v8bf ar_lo = *(const v8bf*)(zr_row + k0);
      const v8bf ar_hi = *(const v8bf*)(zr_row + k0 + 16);
      const v8bf ai_lo = *(const v8bf*)(zi_row + k0);
      const v8bf ai_hi = *(const v8bf*)(zi_row + k0 + 16);
      const v16bf a_r = __builtin_shufflevector(ar_lo, ar_hi,
          0, 1, 2, 3, 4, 5, 6, 7, 8, 9, 10, 11, 12, 13, 14, 15);
      const v16bf a_i = __builtin_shufflevector(ai_lo, ai_hi,
          0, 1, 2, 3, 4, 5, 6, 7, 8, 9, 10, 11, 12, 13, 14, 15);
      const size_t wb = ((((size_t)kb << 6) + tileN) << 9) + (lane << 4);
      const v16bf wr = *(const v16bf*)(pAr + wb);
      const v16bf wi = *(const v16bf*)(pAi + wb);
      rr = __builtin_amdgcn_wmma_f32_16x16x32_bf16(false, a_r, false, wr, (short)0, rr, false, false);
      ii = __builtin_amdgcn_wmma_f32_16x16x32_bf16(false, a_i, false, wi, (short)0, ii, false, false);
      ri = __builtin_amdgcn_wmma_f32_16x16x32_bf16(false, a_r, false, wi, (short)0, ri, false, false);
      ir = __builtin_amdgcn_wmma_f32_16x16x32_bf16(false, a_i, false, wr, (short)0, ir, false, false);
    }

    // ---- input term: x_t @ Bw  (f32 load + pack-convert, off critical path) ----
    for (int kb = 0; kb < 32; ++kb) {
      const int k0 = (kb << 5) + koff;
      v16bf a_r, a_i;
#pragma unroll
      for (int j = 0; j < 8; ++j) {
        a_r[j]     = (__bf16)xr_row[k0 + j];
        a_r[j + 8] = (__bf16)xr_row[k0 + 16 + j];
        a_i[j]     = (__bf16)xi_row[k0 + j];
        a_i[j + 8] = (__bf16)xi_row[k0 + 16 + j];
      }
      const size_t wb = ((((size_t)kb << 6) + tileN) << 9) + (lane << 4);
      const v16bf wr = *(const v16bf*)(pBr + wb);
      const v16bf wi = *(const v16bf*)(pBi + wb);
      rr = __builtin_amdgcn_wmma_f32_16x16x32_bf16(false, a_r, false, wr, (short)0, rr, false, false);
      ii = __builtin_amdgcn_wmma_f32_16x16x32_bf16(false, a_i, false, wi, (short)0, ii, false, false);
      ri = __builtin_amdgcn_wmma_f32_16x16x32_bf16(false, a_r, false, wi, (short)0, ri, false, false);
      ir = __builtin_amdgcn_wmma_f32_16x16x32_bf16(false, a_i, false, wr, (short)0, ir, false, false);
    }

    // ---- epilogue: combine, Hopf on f32 accumulators, store y (f32) + z (bf16) ----
#pragma unroll
    for (int v = 0; v < 8; ++v) {
      const int m = (tileM << 4) + v + ((lane >> 4) << 3);  // batch index
      const int n = (tileN << 4) + (lane & 15);             // unit index
      const float zr_n = 0.5f * (rr[v] - ii[v]);
      const float zi_n = 0.5f * (ri[v] + ir[v]);
      zr_out[(size_t)m * kU + n] = (__bf16)zr_n;
      zi_out[(size_t)m * kU + n] = (__bf16)zi_n;
      const float r2 = zr_n * zr_n + zi_n * zi_n;
      const float g  = 1.0f + kMu - r2;                     // 1 + (mu - |z|^2)
      const float yr = g * zr_n - kOmega * zi_n;
      const float yi = g * zi_n + kOmega * zr_n;
      const size_t o = (((size_t)m * kT + t) * kU + n) << 1;
      out[o]     = yr;
      out[o + 1] = yi;
    }

    // ---- inter-step grid barrier (skip after the final step) ----
    if (t + 1 < kT) {
      __threadfence();        // make this block's z stores device-visible
      __syncthreads();        // all waves in block done + fenced
      if (threadIdx.x == 0) {
        __hip_atomic_fetch_add(bar, 1u, __ATOMIC_RELEASE, __HIP_MEMORY_SCOPE_AGENT);
        const unsigned target = (unsigned)kBlocks * (unsigned)(t + 1);
        while (__hip_atomic_load(bar, __ATOMIC_ACQUIRE, __HIP_MEMORY_SCOPE_AGENT) < target) {
          __builtin_amdgcn_s_sleep(8);
        }
      }
      __syncthreads();        // release whole block into step t+1
    }
  }
}

extern "C" void kernel_launch(void* const* d_in, const int* in_sizes, int n_in,
                              void* d_out, int out_size, void* d_ws, size_t ws_size,
                              hipStream_t stream) {
  const float* xr  = (const float*)d_in[0];
  const float* xi  = (const float*)d_in[1];
  const float* Ar  = (const float*)d_in[2];
  const float* Ai  = (const float*)d_in[3];
  const float* Br  = (const float*)d_in[4];
  const float* Bi  = (const float*)d_in[5];
  const float* z0r = (const float*)d_in[6];
  const float* z0i = (const float*)d_in[7];
  float* out = (float*)d_out;

  // Workspace layout:
  //   [0, 8MB)       : packed bf16 weights [Ar, Ai, Br, Bi], WMMA B-operand layout
  //   [8MB, 8.5MB)   : bf16 state ping-pong: z[2 buffers][re/im][64*1024]
  //   [9MB, 9MB+4)   : inter-step barrier counter
  __bf16*   packed = (__bf16*)d_ws;
  __bf16*   zst    = (__bf16*)((char*)d_ws + ((size_t)8 << 20));
  unsigned* bar    = (unsigned*)((char*)d_ws + ((size_t)9 << 20));

  pack_weights_kernel<<<1024, 256, 0, stream>>>(Ar, Ai, Br, Bi, packed);
  init_state_kernel<<<(kB * kU) / 256, 256, 0, stream>>>(z0r, z0i, zst, zst + kB * kU, bar);
  hopf_persistent_kernel<<<kBlocks, 256, 0, stream>>>(xr, xi, packed, zst, out, bar);
}
